// SimplifiedVirusHostModel_66168266162279
// MI455X (gfx1250) — matrix operand order repaired and checked by reading, not compiled, not run on hardware
//
#include <hip/hip_runtime.h>
#include <hip/hip_bf16.h>

// ---------------------------------------------------------------------------
// CDNA5 (gfx1250) implementation of SimplifiedVirusHostModel forward pass.
// All GEMMs use V_WMMA_F32_16X16X32_BF16 with split-bf16 (hi/lo) operands,
// 3 WMMAs per K-tile, for near-f32 accuracy at bf16 matrix-pipe throughput.
// ---------------------------------------------------------------------------

typedef __attribute__((ext_vector_type(16))) __bf16 v16bf;
typedef __attribute__((ext_vector_type(8)))  __bf16 v8bf;
typedef __attribute__((ext_vector_type(8)))  float  v8f;

#define NPW 8  // 16-col n-tiles per wave -> 128 output columns per wave

__device__ __forceinline__ v16bf load_bfrag(const __bf16* p) {
  v8bf b0 = *reinterpret_cast<const v8bf*>(p);
  v8bf b1 = *reinterpret_cast<const v8bf*>(p + 16);
  return __builtin_shufflevector(b0, b1, 0,1,2,3,4,5,6,7,8,9,10,11,12,13,14,15);
}

// ----------------------- generic bf16x3 WMMA GEMM --------------------------
// A: [Mtiles*16, K] f32 row-major.  B (weights): col-major bf16 hi/lo [N][K].
// C: [Mtiles*16, N] f32 row-major.  Wave computes 16 rows x 128 cols.
__global__ void __launch_bounds__(32)
gemm_bf16x3(const float* __restrict__ A, const __bf16* __restrict__ Bhi,
            const __bf16* __restrict__ Blo, const float* __restrict__ bias,
            float* __restrict__ C, int K, int N, int ktiles, int relu)
{
  const int lane = threadIdx.x & 31;
  const int hv   = lane >> 4;          // which K-half of the fragment
  const int r    = lane & 15;
  const int row  = blockIdx.x * 16 + r;      // A row supplied by this lane
  const int colbase = blockIdx.y * (NPW * 16);

  const float* arow = A + (size_t)row * K;

  const v8f zv = {0.f,0.f,0.f,0.f,0.f,0.f,0.f,0.f};
  v8f acc[NPW];
#pragma unroll
  for (int i = 0; i < NPW; ++i) acc[i] = zv;

  for (int kt = 0; kt < ktiles; ++kt) {
    const int kb = kt * 32 + hv * 8;
    if (kt + 1 < ktiles) __builtin_prefetch(arow + kb + 32, 0, 0);
    // A fragment per ISA layout: lane holds row (lane&15),
    // K = kt*32 + hv*8 + [0..7]  and  kt*32 + 16 + hv*8 + [0..7]
    float4 a0 = *reinterpret_cast<const float4*>(arow + kb);
    float4 a1 = *reinterpret_cast<const float4*>(arow + kb + 4);
    float4 a2 = *reinterpret_cast<const float4*>(arow + kb + 16);
    float4 a3 = *reinterpret_cast<const float4*>(arow + kb + 20);
    float af[16] = {a0.x,a0.y,a0.z,a0.w, a1.x,a1.y,a1.z,a1.w,
                    a2.x,a2.y,a2.z,a2.w, a3.x,a3.y,a3.z,a3.w};
    v16bf ahi, alo;
#pragma unroll
    for (int i = 0; i < 16; ++i) {
      __bf16 h = (__bf16)af[i];
      ahi[i] = h;
      alo[i] = (__bf16)(af[i] - (float)h);
    }
#pragma unroll
    for (int nt = 0; nt < NPW; ++nt) {
      const int col = colbase + nt * 16 + r;
      const size_t bo = (size_t)col * K + kt * 32 + hv * 8; // K = hv*8+[0..7], 16+hv*8+[0..7]
      v16bf bhi = load_bfrag(Bhi + bo);
      v16bf blo = load_bfrag(Blo + bo);
      acc[nt] = __builtin_amdgcn_wmma_f32_16x16x32_bf16(
          false, ahi, false, bhi, (short)0, acc[nt], false, false);
      acc[nt] = __builtin_amdgcn_wmma_f32_16x16x32_bf16(
          false, alo, false, bhi, (short)0, acc[nt], false, false);
      acc[nt] = __builtin_amdgcn_wmma_f32_16x16x32_bf16(
          false, ahi, false, blo, (short)0, acc[nt], false, false);
    }
  }

#pragma unroll
  for (int nt = 0; nt < NPW; ++nt) {
    const int col = colbase + nt * 16 + r;
    const float bb = bias ? bias[col] : 0.f;
#pragma unroll
    for (int rr = 0; rr < 8; ++rr) {
      const int orow = blockIdx.x * 16 + hv * 8 + rr; // C layout: row r+8*half
      float v = acc[nt][rr] + bb;
      if (relu) v = fmaxf(v, 0.f);
      C[(size_t)orow * N + col] = v;
    }
  }
}

// ------------------- weight conversion: f32 -> col-major bf16 hi/lo --------
__global__ void prep_weight(const float* __restrict__ W, __bf16* __restrict__ hi,
                            __bf16* __restrict__ lo, int K, int Nmat)
{
  int idx = blockIdx.x * blockDim.x + threadIdx.x;
  if (idx >= K * Nmat) return;
  int k = idx / Nmat, n = idx - k * Nmat;     // W is [K][Nmat] row-major
  float w = W[idx];
  __bf16 h = (__bf16)w;
  hi[(size_t)n * K + k] = h;
  lo[(size_t)n * K + k] = (__bf16)(w - (float)h);
}

__global__ void zero_f32(float* __restrict__ p, int n)
{
  int i = blockIdx.x * blockDim.x + threadIdx.x;
  if (i < n) p[i] = 0.f;
}

// ------------------------------- GCN -------------------------------------
__global__ void deg_count(const int* __restrict__ ei, float* __restrict__ deg, int E)
{
  int e = blockIdx.x * blockDim.x + threadIdx.x;
  if (e < E) atomicAdd(&deg[ei[E + e]], 1.f);
}

__global__ void deg_rsqrt(float* __restrict__ deg, int n)
{
  int i = blockIdx.x * blockDim.x + threadIdx.x;
  if (i < n) deg[i] = rsqrtf(deg[i] + 1.f);   // +1 for the self-loop
}

// 32 lanes per edge, float4 per lane -> coalesced gather + f32 L2 atomics
__global__ void __launch_bounds__(256)
gcn_scatter(const int* __restrict__ ei, const float* __restrict__ dinv,
            const float* __restrict__ xw, float* __restrict__ agg, int E)
{
  long long t = (long long)blockIdx.x * blockDim.x + threadIdx.x;
  int e = (int)(t >> 5);
  if (e >= E) return;
  int lane = (int)(t & 31);
  int src = ei[e], dst = ei[E + e];
  float c = dinv[src] * dinv[dst];
  const float4 v = *reinterpret_cast<const float4*>(xw + (size_t)src * 128 + lane * 4);
  float* ap = agg + (size_t)dst * 128 + lane * 4;
  atomicAdd(ap + 0, v.x * c);
  atomicAdd(ap + 1, v.y * c);
  atomicAdd(ap + 2, v.z * c);
  atomicAdd(ap + 3, v.w * c);
}

// combine (agg + xw*dinv^2 + b) with adaptive avg pool over nper nodes,
// writing into x[:, coloff:coloff+128]
__global__ void gcn_pool(const float* __restrict__ agg, const float* __restrict__ xw,
                         const float* __restrict__ dinv, const float* __restrict__ bias,
                         float* __restrict__ x, int coloff, int nper)
{
  int idx = blockIdx.x * blockDim.x + threadIdx.x;
  if (idx >= 1000 * 128) return;
  int p = idx >> 7, h = idx & 127;
  float s = 0.f;
  int base = p * nper;
  for (int i = 0; i < nper; ++i) {
    int node = base + i;
    float di = dinv[node];
    s += agg[(size_t)node * 128 + h] + xw[(size_t)node * 128 + h] * di * di;
  }
  x[p * 256 + coloff + h] = s / (float)nper + bias[h];
}

// ---------------------------- attention -----------------------------------
// One wave per (query, head); online softmax per lane over strided keys,
// merged through LDS. Seq len 1000, head dim 32, qkv row = [q|k|v] (768).
__global__ void __launch_bounds__(32)
attention(const float* __restrict__ qkv, float* __restrict__ O)
{
  const int q = blockIdx.x, h = blockIdx.y, lane = threadIdx.x;
  float* orow = O + (size_t)q * 256 + h * 32;
  if (q >= 1000) { orow[lane] = 0.f; return; }   // keep pad rows defined

  const float scale = 0.17677669529663689f;      // 1/sqrt(32)
  float qv[32];
  const float* qp = qkv + (size_t)q * 768 + h * 32;
#pragma unroll
  for (int d = 0; d < 32; ++d) qv[d] = qp[d] * scale;

  float m = -3.0e38f, l = 0.f, o[32];
#pragma unroll
  for (int d = 0; d < 32; ++d) o[d] = 0.f;

  for (int j = lane; j < 1000; j += 32) {
    const float* kp = qkv + (size_t)j * 768 + 256 + h * 32;
    float s = 0.f;
#pragma unroll
    for (int d = 0; d < 32; ++d) s += qv[d] * kp[d];
    float mn = fmaxf(m, s);
    float c  = __expf(m - mn);
    float p  = __expf(s - mn);
    l = l * c + p;
    const float* vp = qkv + (size_t)j * 768 + 512 + h * 32;
#pragma unroll
    for (int d = 0; d < 32; ++d) o[d] = o[d] * c + p * vp[d];
    m = mn;
  }

  __shared__ float s_o[32][33];
  __shared__ float s_m[32], s_l[32];
#pragma unroll
  for (int d = 0; d < 32; ++d) s_o[lane][d] = o[d];
  s_m[lane] = m; s_l[lane] = l;
  __syncthreads();

  float M = m;
  for (int off = 16; off; off >>= 1) M = fmaxf(M, __shfl_xor(M, off));
  float L = 0.f, num = 0.f;
  for (int i = 0; i < 32; ++i) {
    float c = __expf(s_m[i] - M);
    L   += s_l[i] * c;
    num += s_o[i][lane] * c;
  }
  orow[lane] = num / L;
}

// ----------------------- residual add + LayerNorm --------------------------
__global__ void __launch_bounds__(256)
add_ln(float* __restrict__ x, const float* __restrict__ res,
       const float* __restrict__ w, const float* __restrict__ b)
{
  const int row = blockIdx.x, tid = threadIdx.x;
  float* xr = x + (size_t)row * 256;
  if (row >= 1000) { xr[tid] = 0.f; return; }   // keep pad rows zero

  float v = xr[tid] + res[(size_t)row * 256 + tid];
  __shared__ float sm[8];
  float s = v;
  for (int off = 16; off; off >>= 1) s += __shfl_xor(s, off);
  if ((tid & 31) == 0) sm[tid >> 5] = s;
  __syncthreads();
  float tot = 0.f;
#pragma unroll
  for (int i = 0; i < 8; ++i) tot += sm[i];
  const float mean = tot * (1.f / 256.f);
  __syncthreads();
  const float d = v - mean;
  s = d * d;
  for (int off = 16; off; off >>= 1) s += __shfl_xor(s, off);
  if ((tid & 31) == 0) sm[tid >> 5] = s;
  __syncthreads();
  tot = 0.f;
#pragma unroll
  for (int i = 0; i < 8; ++i) tot += sm[i];
  const float var = tot * (1.f / 256.f);
  xr[tid] = d * rsqrtf(var + 1e-5f) * w[tid] + b[tid];
}

// -------------------------- final tiny GEMM (N=2) --------------------------
__global__ void mlp_out(const float* __restrict__ m1, const float* __restrict__ Wm2,
                        const float* __restrict__ bm2, float* __restrict__ out)
{
  int t = blockIdx.x * blockDim.x + threadIdx.x;
  if (t >= 2000) return;
  int p = t >> 1, o = t & 1;
  const float* row = m1 + (size_t)p * 512;
  float s = 0.f;
  for (int k = 0; k < 512; ++k) s += row[k] * Wm2[k * 2 + o];
  out[p * 2 + o] = s + bm2[o];
}

// ---------------------------------------------------------------------------
extern "C" void kernel_launch(void* const* d_in, const int* in_sizes, int n_in,
                              void* d_out, int out_size, void* d_ws, size_t ws_size,
                              hipStream_t stream)
{
  (void)n_in; (void)out_size; (void)ws_size;

  const float* virus = (const float*)d_in[0];
  const float* hostd = (const float*)d_in[1];
  const int*   vei   = (const int*)d_in[3];
  const int*   hei   = (const int*)d_in[4];
  const float* W_gv   = (const float*)d_in[6];
  const float* b_gv   = (const float*)d_in[7];
  const float* W_gh   = (const float*)d_in[8];
  const float* b_gh   = (const float*)d_in[9];
  const float* W_gcnv = (const float*)d_in[12];
  const float* b_gcnv = (const float*)d_in[13];
  const float* W_gcnh = (const float*)d_in[14];
  const float* b_gcnh = (const float*)d_in[15];

  struct Layer { const float *Wqkv,*bqkv,*Wo,*bo,*ln1w,*ln1b,*W1,*b1,*W2,*b2,*ln2w,*ln2b; };
  Layer L[2];
  for (int l = 0; l < 2; ++l) {
    int b = 16 + l * 12;
    L[l].Wqkv=(const float*)d_in[b+0];  L[l].bqkv=(const float*)d_in[b+1];
    L[l].Wo  =(const float*)d_in[b+2];  L[l].bo  =(const float*)d_in[b+3];
    L[l].ln1w=(const float*)d_in[b+4];  L[l].ln1b=(const float*)d_in[b+5];
    L[l].W1  =(const float*)d_in[b+6];  L[l].b1  =(const float*)d_in[b+7];
    L[l].W2  =(const float*)d_in[b+8];  L[l].b2  =(const float*)d_in[b+9];
    L[l].ln2w=(const float*)d_in[b+10]; L[l].ln2b=(const float*)d_in[b+11];
  }
  const float* Wm1 = (const float*)d_in[40];
  const float* bm1 = (const float*)d_in[41];
  const float* Wm2 = (const float*)d_in[42];
  const float* bm2 = (const float*)d_in[43];

  const int Nn   = in_sizes[0] / 384;   // 100000 nodes
  const int E    = in_sizes[3] / 2;     // 800000 edges
  const int Mt   = Nn / 16;             // row tiles for node GEMMs
  const int nper = Nn / 1000;           // pool block size

  // ----- workspace bump allocator -----
  char* wp = (char*)d_ws;
  auto alloc = [&](size_t bytes) -> char* {
    char* p = wp; wp += (bytes + 255) & ~(size_t)255; return p;
  };
  float* bufA  = (float*)alloc((size_t)Nn * 128 * 4); // lin out, then reused as agg
  float* bufXW = (float*)alloc((size_t)Nn * 128 * 4); // gcn xw
  float* dinv  = (float*)alloc((size_t)Nn * 4);
  float* x     = (float*)alloc(1008ull * 256 * 4);
  float* qkvb  = (float*)alloc(1008ull * 768 * 4);
  float* Ob    = (float*)alloc(1008ull * 256 * 4);
  float* t0    = (float*)alloc(1008ull * 256 * 4);
  float* ffb   = (float*)alloc(1008ull * 2048 * 4);
  float* m1b   = (float*)alloc(1008ull * 512 * 4);

  struct BW { __bf16 *hi, *lo; };
  auto prep = [&](const float* W, int K, int Nmat) -> BW {
    BW w;
    w.hi = (__bf16*)alloc((size_t)K * Nmat * 2);
    w.lo = (__bf16*)alloc((size_t)K * Nmat * 2);
    int n = K * Nmat;
    prep_weight<<<(n + 255) / 256, 256, 0, stream>>>(W, w.hi, w.lo, K, Nmat);
    return w;
  };

  BW wgv = prep(W_gv, 384, 128),  wgh = prep(W_gh, 384, 128);
  BW wcv = prep(W_gcnv, 128, 128), wch = prep(W_gcnh, 128, 128);
  BW wqkv[2], wo[2], w1[2], w2[2];
  for (int l = 0; l < 2; ++l) {
    wqkv[l] = prep(L[l].Wqkv, 256, 768);
    wo[l]   = prep(L[l].Wo,   256, 256);
    w1[l]   = prep(L[l].W1,   256, 2048);
    w2[l]   = prep(L[l].W2,  2048, 256);
  }
  BW wm1 = prep(Wm1, 256, 512);

  auto gemm = [&](const float* A, BW B, const float* bias, float* C,
                  int Mtiles, int K, int N, int relu) {
    gemm_bf16x3<<<dim3(Mtiles, N / 128), dim3(32), 0, stream>>>(
        A, B.hi, B.lo, bias, C, K, N, K / 32, relu);
  };
  auto zero = [&](float* p, int n) {
    zero_f32<<<(n + 255) / 256, 256, 0, stream>>>(p, n);
  };

  // ----- per-graph: linear -> GCN GEMM -> degree -> scatter -> pool -----
  auto run_graph = [&](const float* data, const int* ei, BW wlin, const float* blin,
                       BW wgcn, const float* bgcn, int coloff) {
    gemm(data, wlin, blin, bufA, Mt, 384, 128, 0);       // feat = data@W + b
    gemm(bufA, wgcn, nullptr, bufXW, Mt, 128, 128, 0);   // xw = feat@Wgcn
    zero(bufA, Nn * 128);                                // bufA reused as agg
    zero(dinv, Nn);
    deg_count<<<(E + 255) / 256, 256, 0, stream>>>(ei, dinv, E);
    deg_rsqrt<<<(Nn + 255) / 256, 256, 0, stream>>>(dinv, Nn);
    long long st = (long long)E * 32;
    gcn_scatter<<<(unsigned)((st + 255) / 256), 256, 0, stream>>>(ei, dinv, bufXW, bufA, E);
    gcn_pool<<<(1000 * 128 + 255) / 256, 256, 0, stream>>>(bufA, bufXW, dinv, bgcn,
                                                           x, coloff, nper);
  };
  run_graph(virus, vei, wgv, b_gv, wcv, b_gcnv, 0);
  run_graph(hostd, hei, wgh, b_gh, wch, b_gcnh, 128);
  zero(x + 1000 * 256, 8 * 256);                         // zero the M-pad rows

  // ----- transformer encoder layers (M padded 1000 -> 1008 = 63 tiles) -----
  for (int l = 0; l < 2; ++l) {
    gemm(x, wqkv[l], L[l].bqkv, qkvb, 63, 256, 768, 0);
    attention<<<dim3(1008, 8), dim3(32), 0, stream>>>(qkvb, Ob);
    gemm(Ob, wo[l], L[l].bo, t0, 63, 256, 256, 0);
    add_ln<<<1008, 256, 0, stream>>>(x, t0, L[l].ln1w, L[l].ln1b);
    gemm(x, w1[l], L[l].b1, ffb, 63, 256, 2048, 1);      // relu
    gemm(ffb, w2[l], L[l].b2, t0, 63, 2048, 256, 0);
    add_ln<<<1008, 256, 0, stream>>>(x, t0, L[l].ln2w, L[l].ln2b);
  }

  // ----- output MLP -----
  gemm(x, wm1, bm1, m1b, 63, 256, 512, 1);               // relu
  mlp_out<<<(2000 + 255) / 256, 256, 0, stream>>>(m1b, Wm2, bm2, (float*)d_out);
}
